// QkvAttnDecoder_31095563223459
// MI455X (gfx1250) — compile-verified, matrix operand order checked by reading, not run on hardware
//
#include <hip/hip_runtime.h>

// ---------------------------------------------------------------------------
// Problem constants (from reference)
// ---------------------------------------------------------------------------
#define BB      64
#define T_ENC   512
#define T_DEC   512
#define HH      512
#define V1      130
#define V2      50
#define NPROB   (V1 + V2)        // 180
#define NPAD    192              // 180 padded to 64-multiple
#define ROWS    (BB * T_DEC)     // 32768
#define EROWS   (BB * T_ENC)     // 32768
#define EMB_TOT 256

typedef __attribute__((ext_vector_type(16))) __bf16 v16bf;
typedef __attribute__((ext_vector_type(8)))  __bf16 v8bf;
typedef __attribute__((ext_vector_type(8)))  float  v8f;

__device__ __forceinline__ float sigmoidf_(float x) { return 1.0f / (1.0f + __expf(-x)); }

// gfx1250 async global->LDS copy (16B per lane), tracked by ASYNCcnt.
__device__ __forceinline__ void async_load16(const void* gptr, void* lptr) {
    unsigned l = (unsigned)(uintptr_t)lptr;   // low 32 bits of generic addr = LDS offset
    asm volatile("global_load_async_to_lds_b128 %0, %1, off"
                 :: "v"(l), "v"(gptr) : "memory");
}
__device__ __forceinline__ void async_wait0() {
    asm volatile("s_wait_asynccnt 0x0" ::: "memory");
}

// ---------------------------------------------------------------------------
// Elementwise helpers
// ---------------------------------------------------------------------------
__global__ __launch_bounds__(256) void f2bf_kernel(const float* __restrict__ s,
                                                   __bf16* __restrict__ d, long n) {
    long i = (long)blockIdx.x * 256 + threadIdx.x;
    if (i < n) d[i] = (__bf16)s[i];
}

// strided convert: d[row*ld + col] = s[row*cols + col]
__global__ __launch_bounds__(256) void f2bf_strided_kernel(const float* __restrict__ s,
                                                           __bf16* __restrict__ d,
                                                           long rows, long cols, long ld) {
    long i = (long)blockIdx.x * 256 + threadIdx.x;
    if (i < rows * cols) {
        long r = i / cols, c = i % cols;
        d[r * ld + c] = (__bf16)s[i];
    }
}

__global__ __launch_bounds__(256) void zero_bf16_kernel(__bf16* __restrict__ d, long n) {
    long i = (long)blockIdx.x * 256 + threadIdx.x;
    if (i < n) d[i] = (__bf16)0.0f;
}

__global__ __launch_bounds__(256) void copy_f32_kernel(const float* __restrict__ s,
                                                       float* __restrict__ d, long n) {
    long i = (long)blockIdx.x * 256 + threadIdx.x;
    if (i < n) d[i] = s[i];
}

// MultiEmbedding gather: one block per (b,t) row, 256 threads = 256 emb cols
__global__ __launch_bounds__(256) void embed_kernel(const int* __restrict__ tgt,
                                                    const float* __restrict__ e0,
                                                    const float* __restrict__ e1,
                                                    const float* __restrict__ e2,
                                                    __bf16* __restrict__ out) {
    long r = blockIdx.x;
    int  c = threadIdx.x;
    const int* tr = tgt + r * 3;
    float v;
    if (c < 64)       v = e0[(long)tr[0] * 64  + c];
    else if (c < 192) v = e1[(long)tr[1] * 128 + (c - 64)];
    else              v = e2[(long)tr[2] * 64  + (c - 192)];
    out[r * EMB_TOT + c] = (__bf16)v;
}

// ---------------------------------------------------------------------------
// Generic bf16 WMMA GEMM (unit K-stride), async double-buffered staging:
//   C[m,n] = sum_k A[m,k]*B[n,k] + bias[n]
//   A elem : A[bz*a_bs + m*a_rs + k]          (k contiguous)
//   B elem : B[bz*b_bs + n*b_rs + k]          (k contiguous)
//   C elem : c_trans ? C[bz*c_bs + n*c_ld + m] : C[bz*c_bs + m*c_ld + n]
// Block: 256 threads (8 waves), 64x64 output tile, K step 32.
// Pipeline: chunk i+1 flows global->LDS via ASYNCcnt while chunk i runs WMMA;
// one workgroup barrier per chunk.
// Fragment packing per CDNA5 ISA 16-bit A/B layouts
// (lanes 0-15: K=0..7,16..23 ; lanes 16-31: K=8..15,24..31).
// ---------------------------------------------------------------------------
__global__ __launch_bounds__(256) void gemm_bf16_kernel(
    const __bf16* __restrict__ A, long a_rs, long a_bs,
    const __bf16* __restrict__ Bm, long b_rs, long b_bs,
    const float* __restrict__ bias,
    void* __restrict__ C, long c_ld, long c_bs, int c_is_bf16, int c_trans,
    int M, int N, int K, int n_real) {
    __shared__ __bf16 sA[2][64][32];
    __shared__ __bf16 sB[2][64][32];

    const int tid  = threadIdx.x;
    const int wave = tid >> 5;
    const int lane = tid & 31;
    const int lrow = lane & 15;
    const int kh   = (lane < 16) ? 0 : 8;     // K half-select per ISA layout
    const int m0   = blockIdx.y * 64;
    const int n0   = blockIdx.x * 64;
    const int tm   = wave & 3;                // 16-row tile within 64
    const int tn0  = (wave >> 2) * 2;         // two 16-col tiles within 64
    const int sr   = tid >> 2;                // staging row 0..63
    const int sk   = (tid & 3) * 8;           // staging K base 0/8/16/24

    A  += (long)blockIdx.z * a_bs;
    Bm += (long)blockIdx.z * b_bs;

    const __bf16* ga = &A[(long)(m0 + sr) * a_rs + sk];
    const __bf16* gb = &Bm[(long)(n0 + sr) * b_rs + sk];

    auto stage = [&](int k0, int buf) {
        async_load16(ga + k0, &sA[buf][sr][sk]);
        async_load16(gb + k0, &sB[buf][sr][sk]);
    };

    v8f acc0 = {}, acc1 = {};
    stage(0, 0);
    int cur = 0;

    for (int k0 = 0; k0 < K; k0 += 32) {
        async_wait0();          // drain chunk i
        __syncthreads();        // buf[cur] visible; buf[cur^1] free
        if (k0 + 32 < K) {
            stage(k0 + 32, cur ^ 1);          // overlaps the WMMAs below
            if (k0 + 64 < K) {                // L2 warm two chunks ahead
                __builtin_prefetch(ga + k0 + 64, 0, 1);
                __builtin_prefetch(gb + k0 + 64, 0, 1);
            }
        }

        v16bf fa, fb0, fb1;
        {
            const int ar = tm * 16 + lrow;
            const int n1 = tn0 * 16 + lrow;
            const int n2 = n1 + 16;
            #pragma unroll
            for (int i = 0; i < 8; i++) {
                fa [i]     = sA[cur][ar][kh + i];
                fa [8 + i] = sA[cur][ar][16 + kh + i];
                fb0[i]     = sB[cur][n1][kh + i];
                fb0[8 + i] = sB[cur][n1][16 + kh + i];
                fb1[i]     = sB[cur][n2][kh + i];
                fb1[8 + i] = sB[cur][n2][16 + kh + i];
            }
        }
        acc0 = __builtin_amdgcn_wmma_f32_16x16x32_bf16(false, fa, false, fb0,
                                                       (short)0, acc0, false, false);
        acc1 = __builtin_amdgcn_wmma_f32_16x16x32_bf16(false, fa, false, fb1,
                                                       (short)0, acc1, false, false);
        cur ^= 1;
    }

    // Epilogue. C/D layout: VGPR r -> M = r (lanes<16) / 8+r (lanes>=16), N = lane%16
    float*  cf = (float*)C  + (long)blockIdx.z * c_bs;
    __bf16* cb = (__bf16*)C + (long)blockIdx.z * c_bs;
    #pragma unroll
    for (int r = 0; r < 8; r++) {
        const int m = m0 + tm * 16 + ((lane < 16) ? r : 8 + r);
        #pragma unroll
        for (int t = 0; t < 2; t++) {
            const int n = n0 + (tn0 + t) * 16 + lrow;
            if (n < n_real) {
                float v = (t == 0 ? acc0[r] : acc1[r]) + (bias ? bias[n] : 0.0f);
                const long idx = c_trans ? ((long)n * c_ld + m) : ((long)m * c_ld + n);
                if (c_is_bf16) cb[idx] = (__bf16)v;
                else           cf[idx] = v;
            }
        }
    }
}

// ---------------------------------------------------------------------------
// One GRU timestep:  gh = h_in @ Whh^T (+bhh), fused with gate math.
// grid = H/64 blocks (x = 64-col slab of H), 256 threads (8 waves).
// Whh slabs double-buffered via async global->LDS (L2-resident bf16 weights);
// h double-buffered via float4 loads + packed bf16 LDS stores.
// ---------------------------------------------------------------------------
__global__ __launch_bounds__(256) void gru_step_kernel(
    const float* __restrict__ gi,       // [B*T, 3H] input-gate preactivations
    int t, int T,
    const float* __restrict__ h_in,     // [B, H]
    float* __restrict__ h_out,          // [B, H]
    const __bf16* __restrict__ Whh,     // [3H, H] bf16
    const float* __restrict__ bhh,      // [3H]
    float* __restrict__ y)              // [B, T, H]
{
    __shared__ __bf16 sH[2][64][32];
    __shared__ __bf16 sW[2][3][64][32];
    __shared__ float  sGH[3][64][64];

    const int tid  = threadIdx.x;
    const int wave = tid >> 5;
    const int lane = tid & 31;
    const int lrow = lane & 15;
    const int kh   = (lane < 16) ? 0 : 8;
    const int n0   = blockIdx.x * 64;
    const int tm   = wave & 3;
    const int tn0  = (wave >> 2) * 2;
    const int sr   = tid >> 2;
    const int sk   = (tid & 3) * 8;

    auto stageW = [&](int k0, int buf) {
        #pragma unroll
        for (int g = 0; g < 3; g++)
            async_load16(&Whh[(long)(g * HH + n0 + sr) * HH + k0 + sk],
                         &sW[buf][g][sr][sk]);
    };
    auto stageH = [&](int k0, int buf) {
        const float4* hp = (const float4*)&h_in[(long)sr * HH + k0 + sk];
        float4 a = hp[0], b = hp[1];
        v8bf hv;
        hv[0] = (__bf16)a.x; hv[1] = (__bf16)a.y;
        hv[2] = (__bf16)a.z; hv[3] = (__bf16)a.w;
        hv[4] = (__bf16)b.x; hv[5] = (__bf16)b.y;
        hv[6] = (__bf16)b.z; hv[7] = (__bf16)b.w;
        *(v8bf*)&sH[buf][sr][sk] = hv;
    };

    v8f acc[3][2] = {};
    stageW(0, 0);
    stageH(0, 0);
    int cur = 0;

    for (int k0 = 0; k0 < HH; k0 += 32) {
        async_wait0();
        __syncthreads();
        if (k0 + 32 < HH) {
            stageW(k0 + 32, cur ^ 1);   // overlaps WMMAs below
            stageH(k0 + 32, cur ^ 1);
        }

        v16bf fa;
        {
            const int ar = tm * 16 + lrow;
            #pragma unroll
            for (int i = 0; i < 8; i++) {
                fa[i]     = sH[cur][ar][kh + i];
                fa[8 + i] = sH[cur][ar][16 + kh + i];
            }
        }
        #pragma unroll
        for (int g = 0; g < 3; g++) {
            #pragma unroll
            for (int tt = 0; tt < 2; tt++) {
                const int nn = (tn0 + tt) * 16 + lrow;
                v16bf fb;
                #pragma unroll
                for (int i = 0; i < 8; i++) {
                    fb[i]     = sW[cur][g][nn][kh + i];
                    fb[8 + i] = sW[cur][g][nn][16 + kh + i];
                }
                acc[g][tt] = __builtin_amdgcn_wmma_f32_16x16x32_bf16(
                    false, fa, false, fb, (short)0, acc[g][tt], false, false);
            }
        }
        cur ^= 1;
    }
    __syncthreads();

    // dump gh into LDS
    #pragma unroll
    for (int g = 0; g < 3; g++)
        #pragma unroll
        for (int tt = 0; tt < 2; tt++)
            #pragma unroll
            for (int r = 0; r < 8; r++) {
                const int m = tm * 16 + ((lane < 16) ? r : 8 + r);
                const int n = (tn0 + tt) * 16 + lrow;
                sGH[g][m][n] = acc[g][tt][r];
            }
    __syncthreads();

    // GRU cell: 64x64 elems, 16 per thread
    for (int i = tid; i < 64 * 64; i += 256) {
        const int b = i >> 6, c = i & 63;
        const long grow = (long)(b * T + t) * (3 * HH);
        const int  col  = n0 + c;
        float i_r = gi[grow + 0 * HH + col];
        float i_z = gi[grow + 1 * HH + col];
        float i_n = gi[grow + 2 * HH + col];
        float h_r = sGH[0][b][c] + bhh[0 * HH + col];
        float h_z = sGH[1][b][c] + bhh[1 * HH + col];
        float h_n = sGH[2][b][c] + bhh[2 * HH + col];
        float hp  = h_in[(long)b * HH + col];
        float r = sigmoidf_(i_r + h_r);
        float z = sigmoidf_(i_z + h_z);
        float n = tanhf(i_n + r * h_n);
        float hn = (1.0f - z) * n + z * hp;
        h_out[(long)b * HH + col] = hn;
        y[(long)(b * T + t) * HH + col] = hn;
    }
}

// ---------------------------------------------------------------------------
// Masked softmax over encoder dim (axis=1) of score[b, s, t], in place,
// plus a TRANSPOSED bf16 copy wT[b, t, s] for the attn GEMM (unit K-stride).
// One wave per (b,t) column.
// ---------------------------------------------------------------------------
__global__ __launch_bounds__(256) void attn_softmax_kernel(
    float* __restrict__ score, __bf16* __restrict__ wtbf,
    const int* __restrict__ src) {
    const int wave = threadIdx.x >> 5, lane = threadIdx.x & 31;
    const int t = blockIdx.x * 8 + wave;
    const int b = blockIdx.y;
    float* col = score + (long)b * T_ENC * T_DEC + t;
    const int* srcb = src + (long)b * T_ENC * 3;
    __bf16* wrow = wtbf + (long)b * T_ENC * T_DEC + (long)t * T_ENC;

    float x[T_ENC / 32];
    float mx = -3.4e38f;
    #pragma unroll
    for (int i = 0; i < T_ENC / 32; i++) {
        const int s = lane + i * 32;
        float v = col[(long)s * T_DEC];
        v = (srcb[s * 3 + 1] != 0) ? v : -1e9f;
        x[i] = v;
        mx = fmaxf(mx, v);
    }
    #pragma unroll
    for (int o = 16; o >= 1; o >>= 1) mx = fmaxf(mx, __shfl_xor(mx, o, 32));
    float sum = 0.0f;
    #pragma unroll
    for (int i = 0; i < T_ENC / 32; i++) { x[i] = __expf(x[i] - mx); sum += x[i]; }
    #pragma unroll
    for (int o = 16; o >= 1; o >>= 1) sum += __shfl_xor(sum, o, 32);
    const float inv = 1.0f / sum;
    #pragma unroll
    for (int i = 0; i < T_ENC / 32; i++) {
        const int s = lane + i * 32;
        const float wv = x[i] * inv;
        col[(long)s * T_DEC] = wv;
        wrow[s] = (__bf16)wv;
    }
}

// ---------------------------------------------------------------------------
// Dual output softmax: per row, softmax over [0,130) and [130,180).
// One wave per row.
// ---------------------------------------------------------------------------
__global__ __launch_bounds__(256) void out_softmax_kernel(
    const float* __restrict__ logits, float* __restrict__ prob) {
    const int wave = threadIdx.x >> 5, lane = threadIdx.x & 31;
    const long row = (long)blockIdx.x * 8 + wave;
    const float* l = logits + row * NPAD;
    float* p = prob + row * NPROB;
    #pragma unroll
    for (int seg = 0; seg < 2; seg++) {
        const int lo = seg ? V1 : 0;
        const int hi = seg ? NPROB : V1;
        float mx = -3.4e38f;
        for (int i = lo + lane; i < hi; i += 32) mx = fmaxf(mx, l[i]);
        #pragma unroll
        for (int o = 16; o >= 1; o >>= 1) mx = fmaxf(mx, __shfl_xor(mx, o, 32));
        float sum = 0.0f;
        for (int i = lo + lane; i < hi; i += 32) sum += __expf(l[i] - mx);
        #pragma unroll
        for (int o = 16; o >= 1; o >>= 1) sum += __shfl_xor(sum, o, 32);
        const float inv = 1.0f / sum;
        for (int i = lo + lane; i < hi; i += 32) p[i] = __expf(l[i] - mx) * inv;
    }
}

// ---------------------------------------------------------------------------
// Host orchestration
// ---------------------------------------------------------------------------
extern "C" void kernel_launch(void* const* d_in, const int* in_sizes, int n_in,
                              void* d_out, int out_size, void* d_ws, size_t ws_size,
                              hipStream_t stream) {
    (void)in_sizes; (void)n_in; (void)out_size; (void)ws_size;

    const int*   src   = (const int*)  d_in[0];
    const int*   tgt   = (const int*)  d_in[1];
    const float* enc_h = (const float*)d_in[2];
    const float* enc_h0= (const float*)d_in[3];
    const float* emb0  = (const float*)d_in[4];
    const float* emb1  = (const float*)d_in[5];
    const float* emb2  = (const float*)d_in[6];
    const float* Wih0  = (const float*)d_in[7];
    const float* Whh0  = (const float*)d_in[8];
    const float* bih0  = (const float*)d_in[9];
    const float* bhh0  = (const float*)d_in[10];
    const float* Wih1  = (const float*)d_in[11];
    const float* Whh1  = (const float*)d_in[12];
    const float* bih1  = (const float*)d_in[13];
    const float* bhh1  = (const float*)d_in[14];
    const float* Wq    = (const float*)d_in[15];
    const float* bq    = (const float*)d_in[16];
    const float* Wk    = (const float*)d_in[17];
    const float* bk    = (const float*)d_in[18];
    const float* Wv    = (const float*)d_in[19];
    const float* bv    = (const float*)d_in[20];
    const float* Wp    = (const float*)d_in[21];
    const float* bp    = (const float*)d_in[22];

    // ---- workspace carve-up -------------------------------------------------
    char* wsb = (char*)d_ws;
    size_t off = 0;
    auto take = [&](size_t bytes) -> char* {
        char* p = wsb + off;
        off = (off + bytes + 255) & ~(size_t)255;
        return p;
    };
    float*  gi     = (float*) take((size_t)ROWS * 3 * HH * 4);  // reused for logits
    __bf16* embbf  = (__bf16*)take((size_t)ROWS * EMB_TOT * 2);
    float*  h1     = (float*) take((size_t)ROWS * HH * 4);
    float*  dec    = (float*) take((size_t)ROWS * HH * 4);
    __bf16* h1bf   = (__bf16*)take((size_t)ROWS * HH * 2);
    __bf16* cat    = (__bf16*)take((size_t)ROWS * 2 * HH * 2);  // [dec | attn] bf16
    __bf16* encbf  = (__bf16*)take((size_t)EROWS * HH * 2);
    __bf16* qbf    = (__bf16*)take((size_t)ROWS * HH * 2);
    __bf16* kbf    = (__bf16*)take((size_t)EROWS * HH * 2);
    __bf16* vtbf   = (__bf16*)take((size_t)BB * HH * T_ENC * 2);   // vT[b,h,s]
    __bf16* wtbf   = (__bf16*)take((size_t)BB * T_DEC * T_ENC * 2);// wT[b,t,s]
    __bf16* Wih0bf = (__bf16*)take((size_t)3 * HH * EMB_TOT * 2);
    __bf16* Whh0bf = (__bf16*)take((size_t)3 * HH * HH * 2);
    __bf16* Wih1bf = (__bf16*)take((size_t)3 * HH * HH * 2);
    __bf16* Whh1bf = (__bf16*)take((size_t)3 * HH * HH * 2);
    __bf16* Wqbf   = (__bf16*)take((size_t)HH * HH * 2);
    __bf16* Wkbf   = (__bf16*)take((size_t)HH * HH * 2);
    __bf16* Wvbf   = (__bf16*)take((size_t)HH * HH * 2);
    __bf16* Wpbf   = (__bf16*)take((size_t)NPAD * 2 * HH * 2);
    float*  hping  = (float*) take((size_t)BB * HH * 4);
    float*  hpong  = (float*) take((size_t)BB * HH * 4);
    float*  logits = gi;                                        // [ROWS, NPAD]

    float* prob  = (float*)d_out;                               // [ROWS, 180]
    float* score = (float*)d_out + (size_t)ROWS * NPROB;        // [B, T_ENC, T_DEC]

    auto f2bf = [&](const float* s, __bf16* d, long n) {
        f2bf_kernel<<<dim3((unsigned)((n + 255) / 256)), 256, 0, stream>>>(s, d, n);
    };
    auto gemm = [&](const __bf16* A, long ars, long abs_,
                    const __bf16* Bm, long brs, long bbs,
                    const float* bias, void* C, long cld, long cbs,
                    int cbf, int ctr,
                    int M, int N, int K, int nreal, int batch) {
        dim3 g((unsigned)(N / 64), (unsigned)(M / 64), (unsigned)batch);
        gemm_bf16_kernel<<<g, 256, 0, stream>>>(A, ars, abs_, Bm, brs, bbs,
                                                bias, C, cld, cbs, cbf, ctr,
                                                M, N, K, nreal);
    };

    // ---- 1. weight / activation converts -----------------------------------
    f2bf(Wih0, Wih0bf, (long)3 * HH * EMB_TOT);
    f2bf(Whh0, Whh0bf, (long)3 * HH * HH);
    f2bf(Wih1, Wih1bf, (long)3 * HH * HH);
    f2bf(Whh1, Whh1bf, (long)3 * HH * HH);
    f2bf(Wq, Wqbf, (long)HH * HH);
    f2bf(Wk, Wkbf, (long)HH * HH);
    f2bf(Wv, Wvbf, (long)HH * HH);
    f2bf(enc_h, encbf, (long)EROWS * HH);
    zero_bf16_kernel<<<dim3((NPAD * 2 * HH + 255) / 256), 256, 0, stream>>>(
        Wpbf, (long)NPAD * 2 * HH);
    f2bf(Wp, Wpbf, (long)NPROB * 2 * HH);

    // ---- 2. embedding gather ------------------------------------------------
    embed_kernel<<<dim3(ROWS), 256, 0, stream>>>(tgt, emb0, emb1, emb2, embbf);

    // ---- 3. layer-0 input gates:  gi = emb @ Wih0^T + bih0 -----------------
    gemm(embbf, EMB_TOT, 0, Wih0bf, EMB_TOT, 0, bih0,
         gi, 3 * HH, 0, 0, 0, ROWS, 3 * HH, EMB_TOT, 3 * HH, 1);

    // ---- 4. layer-0 recurrence ---------------------------------------------
    copy_f32_kernel<<<dim3((BB * HH + 255) / 256), 256, 0, stream>>>(
        enc_h0, hping, (long)BB * HH);
    for (int t = 0; t < T_DEC; t++) {
        const float* hin = (t & 1) ? hpong : hping;
        float*       hout= (t & 1) ? hping : hpong;
        gru_step_kernel<<<dim3(HH / 64), 256, 0, stream>>>(
            gi, t, T_DEC, hin, hout, Whh0bf, bhh0, h1);
    }

    // ---- 5. layer-1 input gates --------------------------------------------
    f2bf(h1, h1bf, (long)ROWS * HH);
    gemm(h1bf, HH, 0, Wih1bf, HH, 0, bih1,
         gi, 3 * HH, 0, 0, 0, ROWS, 3 * HH, HH, 3 * HH, 1);

    // ---- 6. layer-1 recurrence ---------------------------------------------
    copy_f32_kernel<<<dim3((BB * HH + 255) / 256), 256, 0, stream>>>(
        enc_h0 + (size_t)BB * HH, hping, (long)BB * HH);
    for (int t = 0; t < T_DEC; t++) {
        const float* hin = (t & 1) ? hpong : hping;
        float*       hout= (t & 1) ? hping : hpong;
        gru_step_kernel<<<dim3(HH / 64), 256, 0, stream>>>(
            gi, t, T_DEC, hin, hout, Whh1bf, bhh1, dec);
    }

    // ---- 7. attention -------------------------------------------------------
    // dec -> first half of concat (bf16); also A (row stride 2H) for q GEMM
    f2bf_strided_kernel<<<dim3((unsigned)(((long)ROWS * HH + 255) / 256)), 256, 0, stream>>>(
        dec, cat, ROWS, HH, 2 * HH);
    // q = dec @ Wq^T + bq   (bf16 out)
    gemm(cat, 2 * HH, 0, Wqbf, HH, 0, bq,
         qbf, HH, 0, 1, 0, ROWS, HH, HH, HH, 1);
    // k = enc_hidden @ Wk^T + bk   (bf16 out)
    gemm(encbf, HH, 0, Wkbf, HH, 0, bk, kbf, HH, 0, 1, 0, EROWS, HH, HH, HH, 1);
    // vT[b,h,s] = (enc_hidden @ Wv^T + bv)^T   (batched, transposed bf16 out)
    gemm(encbf, HH, (long)T_ENC * HH, Wvbf, HH, 0, bv,
         vtbf, T_ENC, (long)HH * T_ENC, 1, 1, T_ENC, HH, HH, HH, BB);
    // score[b,s,t] = sum_h k[b,s,h] q[b,t,h]   (f32 into d_out w-region)
    gemm(kbf, HH, (long)T_ENC * HH, qbf, HH, (long)T_DEC * HH,
         nullptr, score, T_DEC, (long)T_ENC * T_DEC, 0, 0,
         T_ENC, T_DEC, HH, T_DEC, BB);
    // masked softmax over s (in place) + transposed bf16 copy wT[b,t,s]
    attn_softmax_kernel<<<dim3(T_DEC / 8, BB), 256, 0, stream>>>(score, wtbf, src);
    // attn[b,t,h] = sum_s wT[b,t,s] vT[b,h,s]  -> second half of concat (bf16)
    gemm(wtbf, T_ENC, (long)T_DEC * T_ENC,
         vtbf, T_ENC, (long)HH * T_ENC,
         nullptr, cat + HH, 2 * HH, (long)T_DEC * 2 * HH, 1, 0,
         T_DEC, HH, T_ENC, HH, BB);

    // ---- 8. output projection + dual softmax -------------------------------
    gemm(cat, 2 * HH, 0, Wpbf, 2 * HH, 0, bp,
         logits, NPAD, 0, 0, 0, ROWS, NPAD, 2 * HH, NPROB, 1);
    out_softmax_kernel<<<dim3(ROWS / 8), 256, 0, stream>>>(logits, prob);
}
